// VectorQuantizer_24661702213811
// MI455X (gfx1250) — matrix operand order repaired and checked by reading, not compiled, not run on hardware
//
#include <hip/hip_runtime.h>
#include <hip/hip_bf16.h>

typedef __bf16 bf16_t;
typedef __attribute__((ext_vector_type(16))) __bf16 v16bf;
typedef __attribute__((ext_vector_type(8)))  float  v8f;

#define NUM_EMB 512
#define EMB_DIM 256
#define HW      4096          // 64*64
#define BATCH   16
#define NROWS   (BATCH * HW)            // 65536 latent vectors
#define NELEM   (BATCH * EMB_DIM * HW)  // 16777216 output elements

__device__ __forceinline__ bf16_t f32_to_bf16_rne(float f) {
    unsigned u = __float_as_uint(f);
    u += 0x7FFFu + ((u >> 16) & 1u);            // round-to-nearest-even
    unsigned short us = (unsigned short)(u >> 16);
    return __builtin_bit_cast(bf16_t, us);
}

// ---------------------------------------------------------------------------
// Kernel 1: per-code ||w||^2 (f32) + bf16 copy of the codebook into workspace.
// 512 blocks x 32 threads (one wave per code row). Also zeroes loss accum.
// ---------------------------------------------------------------------------
__global__ __launch_bounds__(32) void vq_prep(const float* __restrict__ w,
                                              unsigned short* __restrict__ wbf,
                                              float* __restrict__ wnorm,
                                              float* __restrict__ accum) {
    int row  = blockIdx.x;
    int lane = threadIdx.x;               // 0..31, each handles 8 channels
    const float* wr = w + row * EMB_DIM + lane * 8;

    float s = 0.0f;
    unsigned short us[8];
#pragma unroll
    for (int i = 0; i < 8; ++i) {
        float f = wr[i];
        s += f * f;
        unsigned u = __float_as_uint(f);
        u += 0x7FFFu + ((u >> 16) & 1u);
        us[i] = (unsigned short)(u >> 16);
    }
    // pack 8 bf16 -> 16B store
    uint4 q;
    q.x = (unsigned)us[0] | ((unsigned)us[1] << 16);
    q.y = (unsigned)us[2] | ((unsigned)us[3] << 16);
    q.z = (unsigned)us[4] | ((unsigned)us[5] << 16);
    q.w = (unsigned)us[6] | ((unsigned)us[7] << 16);
    reinterpret_cast<uint4*>(wbf + row * EMB_DIM)[lane] = q;

#pragma unroll
    for (int m = 16; m >= 1; m >>= 1) s += __shfl_xor(s, m, 32);
    if (lane == 0) wnorm[row] = s;
    if (row == 0 && lane == 0) *accum = 0.0f;   // reset loss accumulator every launch
}

// ---------------------------------------------------------------------------
// Kernel 2: distances via v_wmma_f32_16x16x32_bf16 + fused argmin.
// Block = 256 threads = 8 waves; each wave owns 16 consecutive latent rows.
// A (z strip, 16x256 bf16) resident in 64 VGPRs; B read from bf16 codebook
// (L2-resident). argmin only needs ||w||^2 - 2 z.w (||z||^2 is row-constant).
// ---------------------------------------------------------------------------
__global__ __launch_bounds__(256) void vq_argmin(const float* __restrict__ x,
                                                 const unsigned short* __restrict__ wbf,
                                                 const float* __restrict__ wnorm,
                                                 float* __restrict__ out_idx) {
    const int lane  = threadIdx.x & 31;
    const int wave  = threadIdx.x >> 5;
    const int strip = blockIdx.x * 8 + wave;   // 16-row strip id
    const int n0    = strip * 16;
    const int b     = n0 >> 12;                // n0 / 4096  (strips never cross images)
    const int hw0   = n0 & 4095;
    const int m     = lane & 15;               // row within strip held by this lane
    const int hiHalf = (lane & 16) ? 1 : 0;

    const float* xb = x + (size_t)b * EMB_DIM * HW + hw0 + m;

    // ---- load A strip: 8 chunks of 16x32 bf16, contiguous k = 16*half + j order, same as B ----
    const int kbase = hiHalf ? 16 : 0;         // K offset for this lane group
    v16bf A[8];
#pragma unroll
    for (int k = 0; k < 8; ++k) {
#pragma unroll
        for (int j = 0; j < 16; ++j) {
            int c = k * 32 + kbase + j;
            A[k][j] = f32_to_bf16_rne(xb[(size_t)c * HW]);
        }
    }

    float best[8];
    int   bidx[8];
#pragma unroll
    for (int v = 0; v < 8; ++v) { best[v] = 3.4e38f; bidx[v] = 0; }

    const int bco = hiHalf ? 16 : 0;           // B-operand channel offset

    for (int t = 0; t < 32; ++t) {             // 32 tiles of 16 codes
        const int code = t * 16 + m;           // code column held by this lane
        const unsigned short* wrow = wbf + code * EMB_DIM;

        v8f acc = {0.f, 0.f, 0.f, 0.f, 0.f, 0.f, 0.f, 0.f};
#pragma unroll
        for (int k = 0; k < 8; ++k) {
            v16bf Bv = *reinterpret_cast<const v16bf*>(wrow + k * 32 + bco);
            acc = __builtin_amdgcn_wmma_f32_16x16x32_bf16(
                false, A[k], false, Bv, (short)0, acc, false, false);
        }

        const float wn = wnorm[code];
#pragma unroll
        for (int v = 0; v < 8; ++v) {
            float d = wn - 2.0f * acc[v];      // + ||z||^2 omitted: row-constant
            if (d < best[v]) { best[v] = d; bidx[v] = code; }
        }
    }

    // ---- argmin across the 16 lanes of each half (index tie-break = first) ----
#pragma unroll
    for (int v = 0; v < 8; ++v) {
#pragma unroll
        for (int mm = 1; mm < 16; mm <<= 1) {
            float od = __shfl_xor(best[v], mm, 32);
            int   oi = __shfl_xor(bidx[v], mm, 32);
            if (od < best[v] || (od == best[v] && oi < bidx[v])) {
                best[v] = od; bidx[v] = oi;
            }
        }
    }

    // lanes 0-15 slots hold rows 0..7, lanes 16-31 slots hold rows 8..15
    if (m == 0) {
        int rbase = n0 + (hiHalf ? 8 : 0);
#pragma unroll
        for (int v = 0; v < 8; ++v) out_idx[rbase + v] = (float)bidx[v];
    }
}

// ---------------------------------------------------------------------------
// Kernel 3: gather f32 code vectors into BCHW output + MSE partial sums.
// One thread per output element; g == flat BCHW index of both x and out.
// ---------------------------------------------------------------------------
__global__ __launch_bounds__(256) void vq_gather(const float* __restrict__ x,
                                                 const float* __restrict__ w,
                                                 const float* __restrict__ idxf,
                                                 float* __restrict__ outq,
                                                 float* __restrict__ accum) {
    const size_t g = (size_t)blockIdx.x * 256 + threadIdx.x;
    const int hw = (int)(g & 4095);
    const int c  = (int)((g >> 12) & 255);
    const int b  = (int)(g >> 20);
    const int n  = (b << 12) | hw;

    const int idx = (int)idxf[n];
    const float wv = w[idx * EMB_DIM + c];
    const float xv = x[g];
    outq[g] = wv;                               // quantized_st == quantized (fwd)

    float d = wv - xv;
    float s = d * d;
#pragma unroll
    for (int mm = 16; mm >= 1; mm >>= 1) s += __shfl_xor(s, mm, 32);

    __shared__ float red[8];
    if ((threadIdx.x & 31) == 0) red[threadIdx.x >> 5] = s;
    __syncthreads();
    if (threadIdx.x < 8) {
        float v = red[threadIdx.x];
#pragma unroll
        for (int mm = 4; mm >= 1; mm >>= 1) v += __shfl_xor(v, mm, 32);
        if (threadIdx.x == 0) atomicAdd(accum, v);
    }
}

// ---------------------------------------------------------------------------
// Kernel 4: loss = (1 + COMMITMENT_COST) * MSE   (q_loss == e_loss in fwd)
// ---------------------------------------------------------------------------
__global__ void vq_finalize(const float* __restrict__ accum, float* __restrict__ loss) {
    *loss = 1.25f * (*accum) / (float)NELEM;
}

extern "C" void kernel_launch(void* const* d_in, const int* in_sizes, int n_in,
                              void* d_out, int out_size, void* d_ws, size_t ws_size,
                              hipStream_t stream) {
    const float* x = (const float*)d_in[0];       // [16, 256, 64, 64] f32
    const float* w = (const float*)d_in[1];       // [512, 256] f32

    float* out  = (float*)d_out;
    float* outq = out;                            // 16777216 floats (quantized_st)
    float* loss = out + NELEM;                    // 1 float
    float* oidx = out + NELEM + 1;                // 65536 floats (indices)

    unsigned short* wbf = (unsigned short*)d_ws;                       // 256 KB bf16 codebook
    float* wnorm = (float*)((char*)d_ws + (size_t)NUM_EMB * EMB_DIM * 2); // 2 KB ||w||^2
    float* accum = wnorm + NUM_EMB;                                    // 4 B loss accumulator

    vq_prep   <<<NUM_EMB, 32, 0, stream>>>(w, wbf, wnorm, accum);
    vq_argmin <<<NROWS / 128, 256, 0, stream>>>(x, wbf, wnorm, oidx);
    vq_gather <<<NELEM / 256, 256, 0, stream>>>(x, w, oidx, outq, accum);
    vq_finalize<<<1, 1, 0, stream>>>(accum, loss);
}
